// Block_42657615184260
// MI455X (gfx1250) — compile-verified, matrix operand order checked by reading, not run on hardware
//
#include <hip/hip_runtime.h>

// ---------------------------------------------------------------------------
// Transformer block forward (B=8, N=1024, C=768, H=12, dk=64, DFF=3072)
// bf16 WMMA (v_wmma_f32_16x16x32_bf16) for all matmuls, flash-attention for
// the softmax(QK^T)V stage, fp32 residual stream.
// ---------------------------------------------------------------------------

typedef __attribute__((ext_vector_type(16))) __bf16 v16bf;
typedef __attribute__((ext_vector_type(8)))  __bf16 v8bf;
typedef __attribute__((ext_vector_type(8)))  float  v8f;

__device__ __forceinline__ unsigned short f2bf(float f) {
    union { float f; unsigned u; } v; v.f = f;
    unsigned r = v.u + 0x7FFFu + ((v.u >> 16) & 1u);   // round-to-nearest-even
    return (unsigned short)(r >> 16);
}

// WMMA 16-bit fragment striping: a lane holds K = kb+{0..7} and kb+16+{0..7}.
// Both halves are contiguous 16-byte chunks -> two b128 loads + a shuffle.
__device__ __forceinline__ v16bf load_frag(const unsigned short* base) {
    v8bf lo = *(const v8bf*)(base);
    v8bf hi = *(const v8bf*)(base + 16);
    return __builtin_shufflevector(lo, hi, 0, 1, 2, 3, 4, 5, 6, 7,
                                           8, 9, 10, 11, 12, 13, 14, 15);
}

__device__ __forceinline__ v8f wmma_bf16(v16bf a, v16bf b, v8f c) {
    return __builtin_amdgcn_wmma_f32_16x16x32_bf16(false, a, false, b,
                                                   (short)0, c, false, false);
}

// ---------------------------------------------------------------------------
__global__ void f32_to_bf16(const float* __restrict__ src,
                            unsigned short* __restrict__ dst, int n) {
    int i = blockIdx.x * blockDim.x + threadIdx.x;
    if (i < n) dst[i] = f2bf(src[i]);
}

// One block per row; cols = 768.
__global__ __launch_bounds__(256)
void layernorm_bf16(const float* __restrict__ x, const float* __restrict__ g,
                    const float* __restrict__ b, unsigned short* __restrict__ out,
                    int cols) {
    int row = blockIdx.x;
    const float* xr = x + (size_t)row * cols;
    __shared__ float ssum[256], ssq[256];
    float s = 0.f, sq = 0.f;
    for (int c = threadIdx.x; c < cols; c += 256) {
        float v = xr[c]; s += v; sq += v * v;
    }
    ssum[threadIdx.x] = s; ssq[threadIdx.x] = sq;
    __syncthreads();
    for (int st = 128; st > 0; st >>= 1) {
        if ((int)threadIdx.x < st) {
            ssum[threadIdx.x] += ssum[threadIdx.x + st];
            ssq[threadIdx.x]  += ssq[threadIdx.x + st];
        }
        __syncthreads();
    }
    float mean = ssum[0] / cols;
    float var  = ssq[0] / cols - mean * mean;
    float rstd = rsqrtf(var + 1e-5f);
    for (int c = threadIdx.x; c < cols; c += 256) {
        float v = (xr[c] - mean) * rstd * g[c] + b[c];
        out[(size_t)row * cols + c] = f2bf(v);
    }
}

// ---------------------------------------------------------------------------
// Generic GEMM: Y[M,N] = A[M,K] (bf16) * W[N,K]^T (bf16) + bias, epilogue by MODE.
// One wave -> 16x128 output tile (8 WMMA column tiles per A fragment).
// Block = 4 waves = 64 rows x 128 cols.
// MODE 0: QKV scatter (q[B,H,N,dk], k[B,H,N,dk], vT[B,H,dk,N], all bf16)
// MODE 1: += residual, fp32 out        (attention projection)
// MODE 2: exact GELU, bf16 out         (FC1)
// MODE 3: += residual, fp32 out        (FC2 -> d_out)
// ---------------------------------------------------------------------------
template <int MODE>
__global__ __launch_bounds__(128)
void gemm_bf16(const unsigned short* __restrict__ A,
               const unsigned short* __restrict__ W,
               const float* __restrict__ bias,
               const float* __restrict__ resid,
               float* __restrict__ outf,
               unsigned short* __restrict__ outb,
               unsigned short* __restrict__ qb,
               unsigned short* __restrict__ kbuf,
               unsigned short* __restrict__ vtb,
               int M, int N, int K) {
    const int lane = threadIdx.x & 31;
    const int wv   = threadIdx.x >> 5;
    const int lo16 = lane & 15;
    const int hi   = lane >> 4;
    const int kb   = hi * 8;
    const int row0 = (blockIdx.y * 4 + wv) * 16;
    const int col0 = blockIdx.x * 128;

    v8f acc[8] = {};
    const unsigned short* arow = A + (size_t)(row0 + lo16) * K + kb;
    const unsigned short* wrow = W + (size_t)(col0 + lo16) * K + kb;

    for (int k0 = 0; k0 < K; k0 += 32) {
        v16bf af = load_frag(arow + k0);
#pragma unroll
        for (int t = 0; t < 8; ++t) {
            v16bf bf = load_frag(wrow + (size_t)(t * 16) * K + k0);
            acc[t] = wmma_bf16(af, bf, acc[t]);
        }
    }

#pragma unroll
    for (int t = 0; t < 8; ++t) {
        const int n  = col0 + t * 16 + lo16;
        const float bv = bias[n];
#pragma unroll
        for (int r = 0; r < 8; ++r) {
            const int m = row0 + r + 8 * hi;   // C/D layout: hi half-wave -> M+8
            float val = acc[t][r] + bv;
            if constexpr (MODE == 0) {
                // n in [0,2304): sec 0=Q, 1=K, 2=V; head = (n%768)/64, d = n%64
                int sec  = n / 768;
                int rem  = n - sec * 768;
                int h    = rem >> 6;
                int d    = rem & 63;
                int bidx = m >> 10;
                int tseq = m & 1023;
                size_t bh = (size_t)(bidx * 12 + h);
                unsigned short bvv = f2bf(val);
                if (sec == 0)      qb  [(bh * 1024 + tseq) * 64 + d] = bvv;
                else if (sec == 1) kbuf[(bh * 1024 + tseq) * 64 + d] = bvv;
                else               vtb [(bh * 64 + d) * 1024 + tseq] = bvv;
            } else if constexpr (MODE == 1) {
                outf[(size_t)m * N + n] = resid[(size_t)m * N + n] + val;
            } else if constexpr (MODE == 2) {
                float g = 0.5f * val * (1.0f + erff(val * 0.70710678118654752f));
                outb[(size_t)m * N + n] = f2bf(g);
            } else {
                outf[(size_t)m * N + n] = resid[(size_t)m * N + n] + val;
            }
        }
    }
}

// ---------------------------------------------------------------------------
// Flash attention: one wave handles a 16-row query tile for one (b, h).
// Online softmax over key chunks of 32; P staged through LDS to convert the
// WMMA C/D layout into the A-fragment layout for the P*V WMMA.
// ---------------------------------------------------------------------------
__device__ __forceinline__ float rowmax16(float v) {
#pragma unroll
    for (int m = 1; m < 16; m <<= 1) v = fmaxf(v, __shfl_xor(v, m, 32));
    return v;
}
__device__ __forceinline__ float rowsum16(float v) {
#pragma unroll
    for (int m = 1; m < 16; m <<= 1) v += __shfl_xor(v, m, 32);
    return v;
}

__global__ __launch_bounds__(128)
void flash_attn(const unsigned short* __restrict__ Q,   // [B*H, N, 64]
                const unsigned short* __restrict__ Kb,  // [B*H, N, 64]
                const unsigned short* __restrict__ Vt,  // [B*H, 64, N]
                unsigned short* __restrict__ O,         // [B, N, 768] bf16
                int Nseq) {
    const int lane = threadIdx.x & 31;
    const int wv   = threadIdx.x >> 5;
    const int lo16 = lane & 15;
    const int hi   = lane >> 4;
    const int kb   = hi * 8;
    const int h    = blockIdx.y;
    const int b    = blockIdx.z;
    const int q0   = (blockIdx.x * 4 + wv) * 16;
    const size_t bh = (size_t)(b * 12 + h);

    __shared__ unsigned short plds_all[4][16 * 32];
    unsigned short* plds = plds_all[wv];

    // Q fragments (dk = 64 -> two K=32 fragments)
    const unsigned short* qrow = Q + (bh * Nseq + q0 + lo16) * 64 + kb;
    const v16bf aq0 = load_frag(qrow);
    const v16bf aq1 = load_frag(qrow + 32);

    v8f acc[4] = {};
    float run_max[8], run_sum[8];
#pragma unroll
    for (int r = 0; r < 8; ++r) { run_max[r] = -1e30f; run_sum[r] = 0.f; }

    const float scale = 0.125f;   // dk^-0.5 = 1/8

    for (int j0 = 0; j0 < Nseq; j0 += 32) {
        // ---- S = Q * K^T for 32 keys (two 16x16 D tiles) ----
        const unsigned short* krow0 = Kb + (bh * Nseq + j0 + lo16) * 64 + kb;
        const unsigned short* krow1 = krow0 + 16 * 64;
        v16bf bk00 = load_frag(krow0);
        v16bf bk01 = load_frag(krow0 + 32);
        v16bf bk10 = load_frag(krow1);
        v16bf bk11 = load_frag(krow1 + 32);
        v8f s0 = {}, s1 = {};
        s0 = wmma_bf16(aq0, bk00, s0);
        s0 = wmma_bf16(aq1, bk01, s0);
        s1 = wmma_bf16(aq0, bk10, s1);
        s1 = wmma_bf16(aq1, bk11, s1);

        // ---- online softmax (row stats live in the C/D row-group lanes) ----
#pragma unroll
        for (int r = 0; r < 8; ++r) {
            float a0 = s0[r] * scale;
            float a1 = s1[r] * scale;
            float mx = rowmax16(fmaxf(a0, a1));
            float nm = fmaxf(run_max[r], mx);
            float alpha = __expf(run_max[r] - nm);
            run_max[r] = nm;
            float p0 = __expf(a0 - nm);
            float p1 = __expf(a1 - nm);
            float cs = rowsum16(p0 + p1);
            run_sum[r] = run_sum[r] * alpha + cs;
#pragma unroll
            for (int t = 0; t < 4; ++t) acc[t][r] *= alpha;
            int m = r + 8 * hi;
            plds[m * 32 + lo16]      = f2bf(p0);
            plds[m * 32 + 16 + lo16] = f2bf(p1);
        }
        __syncthreads();

        // ---- reload P as an A fragment (16x32, K = key-in-chunk) ----
        v16bf ap = load_frag(plds + lo16 * 32 + kb);   // ds_load_b128 x2

        // ---- O += P * V (V transposed: contiguous over keys) ----
#pragma unroll
        for (int t = 0; t < 4; ++t) {
            const unsigned short* vrow =
                Vt + (bh * 64 + t * 16 + lo16) * (size_t)Nseq + j0 + kb;
            v16bf bv = load_frag(vrow);
            acc[t] = wmma_bf16(ap, bv, acc[t]);
        }
        __syncthreads();
    }

    // ---- normalize and write out: O[b, m, h*64 + d] ----
#pragma unroll
    for (int t = 0; t < 4; ++t) {
#pragma unroll
        for (int r = 0; r < 8; ++r) {
            int m = q0 + r + 8 * hi;
            float val = acc[t][r] / run_sum[r];
            O[((size_t)b * Nseq + m) * 768 + h * 64 + t * 16 + lo16] = f2bf(val);
        }
    }
}

// ---------------------------------------------------------------------------
extern "C" void kernel_launch(void* const* d_in, const int* in_sizes, int n_in,
                              void* d_out, int out_size, void* d_ws, size_t ws_size,
                              hipStream_t stream) {
    const float* x      = (const float*)d_in[0];
    const float* ln1_w  = (const float*)d_in[1];
    const float* ln1_b  = (const float*)d_in[2];
    const float* qkv_w  = (const float*)d_in[3];
    const float* qkv_b  = (const float*)d_in[4];
    const float* proj_w = (const float*)d_in[5];
    const float* proj_b = (const float*)d_in[6];
    const float* ln2_w  = (const float*)d_in[7];
    const float* ln2_b  = (const float*)d_in[8];
    const float* fc1_w  = (const float*)d_in[9];
    const float* fc1_b  = (const float*)d_in[10];
    const float* fc2_w  = (const float*)d_in[11];
    const float* fc2_b  = (const float*)d_in[12];
    float* out = (float*)d_out;

    const int Bx = 8, Nseq = 1024, C = 768, HEADS = 12, DFF = 3072;
    const int M = Bx * Nseq;                  // 8192 rows

    size_t off = 0;
    auto carve = [&](size_t bytes) -> char* {
        char* p = (char*)d_ws + off;
        off = (off + bytes + 255) & ~(size_t)255;
        return p;
    };
    unsigned short* wqkv  = (unsigned short*)carve((size_t)3 * C * C * 2);
    unsigned short* wproj = (unsigned short*)carve((size_t)C * C * 2);
    unsigned short* wfc1  = (unsigned short*)carve((size_t)DFF * C * 2);
    unsigned short* wfc2  = (unsigned short*)carve((size_t)C * DFF * 2);
    unsigned short* hb    = (unsigned short*)carve((size_t)M * C * 2);
    unsigned short* qb    = (unsigned short*)carve((size_t)M * C * 2);
    unsigned short* kbuf  = (unsigned short*)carve((size_t)M * C * 2);
    unsigned short* vtb   = (unsigned short*)carve((size_t)M * C * 2);
    unsigned short* attnb = (unsigned short*)carve((size_t)M * C * 2);
    float*          x1    = (float*)carve((size_t)M * C * 4);
    unsigned short* hb2   = (unsigned short*)carve((size_t)M * C * 2);
    unsigned short* h2b   = (unsigned short*)carve((size_t)M * DFF * 2);

    // --- weights -> bf16 ---
    int nw;
    nw = 3 * C * C; f32_to_bf16<<<(nw + 255) / 256, 256, 0, stream>>>(qkv_w, wqkv, nw);
    nw = C * C;     f32_to_bf16<<<(nw + 255) / 256, 256, 0, stream>>>(proj_w, wproj, nw);
    nw = DFF * C;   f32_to_bf16<<<(nw + 255) / 256, 256, 0, stream>>>(fc1_w, wfc1, nw);
    nw = C * DFF;   f32_to_bf16<<<(nw + 255) / 256, 256, 0, stream>>>(fc2_w, wfc2, nw);

    // --- LN1 -> bf16 activations ---
    layernorm_bf16<<<M, 256, 0, stream>>>(x, ln1_w, ln1_b, hb, C);

    // --- QKV projection (scatter into per-head Q/K + transposed V) ---
    gemm_bf16<0><<<dim3(3 * C / 128, M / 64), 128, 0, stream>>>(
        hb, wqkv, qkv_b, nullptr, nullptr, nullptr, qb, kbuf, vtb, M, 3 * C, C);

    // --- attention ---
    flash_attn<<<dim3(Nseq / 64, HEADS, Bx), 128, 0, stream>>>(qb, kbuf, vtb, attnb, Nseq);

    // --- output projection + residual -> x1 (fp32) ---
    gemm_bf16<1><<<dim3(C / 128, M / 64), 128, 0, stream>>>(
        attnb, wproj, proj_b, x, x1, nullptr, nullptr, nullptr, nullptr, M, C, C);

    // --- LN2 -> bf16 ---
    layernorm_bf16<<<M, 256, 0, stream>>>(x1, ln2_w, ln2_b, hb2, C);

    // --- FC1 + exact GELU -> bf16 ---
    gemm_bf16<2><<<dim3(DFF / 128, M / 64), 128, 0, stream>>>(
        hb2, wfc1, fc1_b, nullptr, nullptr, h2b, nullptr, nullptr, nullptr, M, DFF, C);

    // --- FC2 + residual -> d_out (fp32) ---
    gemm_bf16<3><<<dim3(C / 128, M / 64), 128, 0, stream>>>(
        h2b, wfc2, fc2_b, x1, out, nullptr, nullptr, nullptr, nullptr, M, C, DFF);
}
